// QFormerCrossAttention_49091476193481
// MI455X (gfx1250) — compile-verified, matrix operand order checked by reading, not run on hardware
//
#include <hip/hip_runtime.h>
#include <hip/hip_bf16.h>

// ---------- types ----------
typedef __attribute__((ext_vector_type(16))) __bf16        v16bf;
typedef __attribute__((ext_vector_type(8)))  float         v8f;
typedef __attribute__((ext_vector_type(4)))  float         f32x4;
typedef __attribute__((ext_vector_type(4)))  unsigned int  u32x4;

union FragU {
  v16bf bf;
  u32x4 q[2];
  __bf16 h[16];
  unsigned short us[16];
};

// fp32 -> bf16 via native conversion (v_cvt_pk_bf16_f32 pairs)
__device__ __forceinline__ unsigned short f2bf(float f) {
  union { __bf16 h; unsigned short u; } c;
  c.h = (__bf16)f;
  return c.u;
}

// 16x32 bf16 A-fragment (also B-fragment of a pre-transposed B) from
// row-major bf16 memory, leading dimension ld (elements).
// ISA layout: lane m = lane&15; lane<16 holds K in [0,8)+[16,24),
// lane>=16 holds K in [8,16)+[24,32).
__device__ __forceinline__ v16bf load_frag_bf16(const unsigned short* base, int ld) {
  const int lane = threadIdx.x & 31;
  const int m    = lane & 15;
  const int ko   = (lane >> 4) << 3;
  const unsigned short* row = base + m * ld + ko;
  FragU f;
  f.q[0] = *(const u32x4*)(row);
  f.q[1] = *(const u32x4*)(row + 16);
  return f.bf;
}

// same fragment but sourced from row-major fp32, converting in registers
__device__ __forceinline__ v16bf load_frag_f32(const float* base, int ld) {
  const int lane = threadIdx.x & 31;
  const int m    = lane & 15;
  const int ko   = (lane >> 4) << 3;
  const float* row = base + m * ld + ko;
  f32x4 a0 = *(const f32x4*)(row);
  f32x4 a1 = *(const f32x4*)(row + 4);
  f32x4 a2 = *(const f32x4*)(row + 16);
  f32x4 a3 = *(const f32x4*)(row + 20);
  FragU f;
#pragma unroll
  for (int i = 0; i < 4; ++i) {
    f.h[i]      = (__bf16)a0[i];
    f.h[4 + i]  = (__bf16)a1[i];
    f.h[8 + i]  = (__bf16)a2[i];
    f.h[12 + i] = (__bf16)a3[i];
  }
  return f.bf;
}

__device__ __forceinline__ v8f wmma_bf16(v16bf a, v16bf b, v8f c) {
  return __builtin_amdgcn_wmma_f32_16x16x32_bf16(false, a, false, b, (short)0, c,
                                                 false, false);
}

// per-lane 16-byte async copy global -> LDS (tracked by ASYNCcnt)
__device__ __forceinline__ void async_copy16(unsigned lds_off, const void* g) {
  asm volatile("global_load_async_to_lds_b128 %0, %1, off"
               :
               : "v"(lds_off), "v"(g)
               : "memory");
}
__device__ __forceinline__ void wait_async0() {
  asm volatile("s_wait_asynccnt 0x0" ::: "memory");
}

// ---------- kernel 1: weight convert + transpose: Wt[n][k] = bf16(W[k][n]) ----------
__global__ void wconv_kernel(const float* __restrict__ W,
                             unsigned short* __restrict__ Wt, int K, int N) {
  int idx = blockIdx.x * blockDim.x + threadIdx.x;
  if (idx >= K * N) return;
  int k = idx / N, n = idx - k * N;
  Wt[(size_t)n * K + k] = f2bf(W[idx]);
}

// ---------- kernel 2: Q projection (scale folded), out (b,h,64,64) bf16 ----------
__global__ void qproj_kernel(const float* __restrict__ query,
                             const unsigned short* __restrict__ Wqt,
                             const float* __restrict__ bq,
                             unsigned short* __restrict__ qbf) {
  const int blk = blockIdx.x;          // 0..191 = (b,h)
  const int b = blk / 12, h = blk % 12;
  const int w = threadIdx.x >> 5;      // 4 waves -> 16-row m strips
  const int lane = threadIdx.x & 31;
  const int hi = lane >> 4, lo = lane & 15;
  const int nbase = h * 64;
  const float* Arow = query + (size_t)(b * 64 + w * 16) * 768;
  v8f acc[4] = {};
  for (int k = 0; k < 768; k += 32) {
    v16bf a = load_frag_f32(Arow + k, 768);
#pragma unroll
    for (int t = 0; t < 4; ++t) {
      v16bf bm = load_frag_bf16(Wqt + (size_t)(nbase + t * 16) * 768 + k, 768);
      acc[t] = wmma_bf16(a, bm, acc[t]);
    }
  }
  unsigned short* qdst = qbf + (size_t)(b * 12 + h) * 64 * 64;
#pragma unroll
  for (int t = 0; t < 4; ++t) {
    const float bias = bq[nbase + t * 16 + lo];
#pragma unroll
    for (int r = 0; r < 8; ++r) {
      int m = w * 16 + r + 8 * hi;
      qdst[m * 64 + t * 16 + lo] = f2bf((acc[t][r] + bias) * 0.125f);  // 1/sqrt(64)
    }
  }
}

// ---------- kernel 3: K/V projection, M=65536 N=768 K=1024 ----------
// 12 waves tile the full N=768 (64 cols each); each wave covers two 16-row
// m-strips; A/B fragments are register double-buffered so the 8-WMMA chain
// overlaps the next k-step's loads. kv streamed from HBM exactly once.
// transposed=0 -> out (b,h,kv,d) ; transposed=1 -> out (b,h,d,kv)
__global__ void __launch_bounds__(384)
kvproj_kernel(const float* __restrict__ kv,
              const unsigned short* __restrict__ Wt,
              const float* __restrict__ bias,
              unsigned short* __restrict__ outp, int transposed) {
  const int mblk = blockIdx.x;       // 32 rows each -> 2048 blocks
  const int wid = threadIdx.x >> 5;  // 0..11 -> n-part
  const int lane = threadIdx.x & 31;
  const int hi = lane >> 4, lo = lane & 15;
  const int nbase = wid * 64;
  const float* Arow0 = kv + (size_t)(mblk * 32) * 1024;
  const float* Arow1 = Arow0 + 16 * 1024;
  const unsigned short* Wrow = Wt + (size_t)nbase * 1024;
  v8f acc[2][4] = {};
  // prologue loads
  v16bf a0 = load_frag_f32(Arow0, 1024);
  v16bf a1 = load_frag_f32(Arow1, 1024);
  v16bf bm[4];
#pragma unroll
  for (int t = 0; t < 4; ++t) bm[t] = load_frag_bf16(Wrow + (size_t)t * 16 * 1024, 1024);
  for (int k = 0; k < 1024; k += 32) {
    v16bf na0, na1, nb[4];
    const int kn = k + 32;
    if (kn < 1024) {
      __builtin_prefetch(Arow0 + (lane & 15) * 1024 + kn + 32, 0, 1);
      __builtin_prefetch(Arow1 + (lane & 15) * 1024 + kn + 32, 0, 1);
      na0 = load_frag_f32(Arow0 + kn, 1024);
      na1 = load_frag_f32(Arow1 + kn, 1024);
#pragma unroll
      for (int t = 0; t < 4; ++t)
        nb[t] = load_frag_bf16(Wrow + (size_t)t * 16 * 1024 + kn, 1024);
    }
#pragma unroll
    for (int t = 0; t < 4; ++t) {
      acc[0][t] = wmma_bf16(a0, bm[t], acc[0][t]);
      acc[1][t] = wmma_bf16(a1, bm[t], acc[1][t]);
    }
    a0 = na0;
    a1 = na1;
#pragma unroll
    for (int t = 0; t < 4; ++t) bm[t] = nb[t];
  }
#pragma unroll
  for (int t = 0; t < 4; ++t) {
    const int n = nbase + t * 16 + lo;
    const float bv = bias[n];
    const int h = n >> 6, d = n & 63;
#pragma unroll
    for (int ms = 0; ms < 2; ++ms) {
#pragma unroll
      for (int r = 0; r < 8; ++r) {
        int mg = mblk * 32 + ms * 16 + r + 8 * hi;
        int b = mg >> 12;
        int kvi = mg & 4095;
        size_t idx = transposed
                         ? ((size_t)((b * 12 + h) * 64 + d) * 4096 + kvi)
                         : ((size_t)((b * 12 + h) * 4096 + kvi) * 64 + d);
        outp[idx] = f2bf(acc[ms][t][r] + bv);
      }
    }
  }
}

// ---------- kernel 4: flash attention per (b,h); out x (b,64,768) bf16 ----------
// K/V chunks are double-buffered into LDS with global_load_async_to_lds_b128
// so the copy of chunk c+1 overlaps compute on chunk c.
__global__ void __launch_bounds__(128)
attn_kernel(const unsigned short* __restrict__ qbf,
            const unsigned short* __restrict__ kbf,
            const unsigned short* __restrict__ vbf,
            unsigned short* __restrict__ xbf) {
  __shared__ __align__(16) unsigned short kbuf[2][64 * 64];  // 2 x 8KB
  __shared__ __align__(16) unsigned short vbuf[2][64 * 64];  // 2 x 8KB
  __shared__ __align__(16) unsigned short pbuf[64 * 64];     // P transpose staging
  const int blk = blockIdx.x;  // (b,h)
  const int b = blk / 12, h = blk % 12;
  const int tid = threadIdx.x;
  const int w = tid >> 5;  // 4 waves x 16-row strips
  const int lane = tid & 31;
  const int hi = lane >> 4, lo = lane & 15;
  const unsigned short* qh = qbf + (size_t)(b * 12 + h) * 64 * 64;
  const unsigned short* kh = kbf + (size_t)(b * 12 + h) * 4096 * 64;
  const unsigned short* vh = vbf + (size_t)(b * 12 + h) * 64 * 4096;

  // issue async copies of one (K chunk, V chunk) pair into LDS buffer bufi
  auto issue_chunk = [&](int c, int bufi) {
    const int kv0 = c * 64;
    // K chunk: rows kv0..kv0+63 of (kv,d)-major K -> one contiguous 8KB block
    const char* kg = (const char*)(kh + (size_t)kv0 * 64);
    unsigned klds = (unsigned)(size_t)&kbuf[bufi][0];
#pragma unroll
    for (int i = 0; i < 4; ++i) {
      int off = tid * 16 + i * 2048;
      async_copy16(klds + off, kg + off);
    }
    // V chunk: 64 rows (d) x 128B from (d,kv)-major V
    unsigned vlds = (unsigned)(size_t)&vbuf[bufi][0];
#pragma unroll
    for (int i = 0; i < 4; ++i) {
      int p = (tid + i * 128) * 16;  // byte position in 64x64 bf16 tile
      int row = p >> 7;              // d row (128B per row)
      int col = p & 127;             // byte within row
      const char* vg = (const char*)(vh + (size_t)row * 4096 + kv0) + col;
      async_copy16(vlds + p, vg);
    }
  };

  // resident Q fragments (16x32 each, K-dim = head dim 64)
  v16bf aq0 = load_frag_bf16(qh + (w * 16) * 64 + 0, 64);
  v16bf aq1 = load_frag_bf16(qh + (w * 16) * 64 + 32, 64);
  v8f o[4] = {};
  float mrow[8], lrow[8];
#pragma unroll
  for (int r = 0; r < 8; ++r) { mrow[r] = -1e30f; lrow[r] = 0.0f; }

  issue_chunk(0, 0);
  for (int c = 0; c < 64; ++c) {
    wait_async0();
    __syncthreads();  // chunk c resident; everyone done reading buffer (c+1)&1
    if (c + 1 < 64) issue_chunk(c + 1, (c + 1) & 1);
    const unsigned short* kc = &kbuf[c & 1][0];
    const unsigned short* vc = &vbuf[c & 1][0];
    // S = Q * K^T from LDS
    v8f s[4] = {};
#pragma unroll
    for (int t = 0; t < 4; ++t) {
      v16bf bk0 = load_frag_bf16(kc + (t * 16) * 64 + 0, 64);
      s[t] = wmma_bf16(aq0, bk0, s[t]);
      v16bf bk1 = load_frag_bf16(kc + (t * 16) * 64 + 32, 64);
      s[t] = wmma_bf16(aq1, bk1, s[t]);
    }
    // online softmax: row m = r + 8*hi lives in one 16-lane half per VGPR r
#pragma unroll
    for (int r = 0; r < 8; ++r) {
      float v = fmaxf(fmaxf(s[0][r], s[1][r]), fmaxf(s[2][r], s[3][r]));
      v = fmaxf(v, __shfl_xor(v, 1, 32));
      v = fmaxf(v, __shfl_xor(v, 2, 32));
      v = fmaxf(v, __shfl_xor(v, 4, 32));
      v = fmaxf(v, __shfl_xor(v, 8, 32));
      float mnew = fmaxf(mrow[r], v);
      float alpha = __expf(mrow[r] - mnew);
      mrow[r] = mnew;
      float rs = 0.0f;
#pragma unroll
      for (int t = 0; t < 4; ++t) {
        float p = __expf(s[t][r] - mnew);
        s[t][r] = p;
        rs += p;
      }
      rs += __shfl_xor(rs, 1, 32);
      rs += __shfl_xor(rs, 2, 32);
      rs += __shfl_xor(rs, 4, 32);
      rs += __shfl_xor(rs, 8, 32);
      lrow[r] = lrow[r] * alpha + rs;
#pragma unroll
      for (int t = 0; t < 4; ++t) o[t][r] *= alpha;
    }
    // C-layout -> A-layout transpose of P through LDS (own strip only)
#pragma unroll
    for (int t = 0; t < 4; ++t)
#pragma unroll
      for (int r = 0; r < 8; ++r)
        pbuf[(w * 16 + r + 8 * hi) * 64 + t * 16 + lo] = f2bf(s[t][r]);
    // O += P * V from LDS
#pragma unroll
    for (int ks = 0; ks < 2; ++ks) {
      v16bf ap = load_frag_bf16(&pbuf[(w * 16) * 64 + ks * 32], 64);
#pragma unroll
      for (int nt = 0; nt < 4; ++nt) {
        v16bf bv = load_frag_bf16(vc + (nt * 16) * 64 + ks * 32, 64);
        o[nt] = wmma_bf16(ap, bv, o[nt]);
      }
    }
  }
  // epilogue: normalize, write x[b, m, h*64 + d] as bf16
  unsigned short* xd = xbf + (size_t)(b * 64) * 768 + h * 64;
#pragma unroll
  for (int nt = 0; nt < 4; ++nt)
#pragma unroll
    for (int r = 0; r < 8; ++r) {
      int m = w * 16 + r + 8 * hi;
      xd[(size_t)m * 768 + nt * 16 + lo] = f2bf(o[nt][r] / lrow[r]);
    }
}

// ---------- kernel 5: output projection, fp32 result ----------
__global__ void oproj_kernel(const unsigned short* __restrict__ xbf,
                             const unsigned short* __restrict__ Wot,
                             const float* __restrict__ bo,
                             float* __restrict__ out) {
  const int blk = blockIdx.x;  // 16 mblocks x 12 nblocks
  const int mb = blk / 12, nb = blk % 12;
  const int w = threadIdx.x >> 5;
  const int lane = threadIdx.x & 31;
  const int hi = lane >> 4, lo = lane & 15;
  const int mstrip = mb * 64 + w * 16;
  const int nbase = nb * 64;
  const unsigned short* Arow = xbf + (size_t)mstrip * 768;
  v8f acc[4] = {};
  for (int k = 0; k < 768; k += 32) {
    v16bf a = load_frag_bf16(Arow + k, 768);
#pragma unroll
    for (int t = 0; t < 4; ++t) {
      v16bf bm = load_frag_bf16(Wot + (size_t)(nbase + t * 16) * 768 + k, 768);
      acc[t] = wmma_bf16(a, bm, acc[t]);
    }
  }
#pragma unroll
  for (int t = 0; t < 4; ++t) {
    const float bias = bo[nbase + t * 16 + lo];
#pragma unroll
    for (int r = 0; r < 8; ++r) {
      int m = mstrip + r + 8 * hi;
      out[(size_t)m * 768 + nbase + t * 16 + lo] = acc[t][r] + bias;
    }
  }
}

extern "C" void kernel_launch(void* const* d_in, const int* in_sizes, int n_in,
                              void* d_out, int out_size, void* d_ws, size_t ws_size,
                              hipStream_t stream) {
  (void)in_sizes; (void)n_in; (void)out_size; (void)ws_size;
  const float* query = (const float*)d_in[0];
  const float* kv    = (const float*)d_in[1];
  const float* Wq    = (const float*)d_in[2];
  const float* bq    = (const float*)d_in[3];
  const float* Wk    = (const float*)d_in[4];
  const float* bk    = (const float*)d_in[5];
  const float* Wv    = (const float*)d_in[6];
  const float* bv    = (const float*)d_in[7];
  const float* Wo    = (const float*)d_in[8];
  const float* bo    = (const float*)d_in[9];
  float* out = (float*)d_out;

  char* ws = (char*)d_ws;
  size_t off = 0;
  auto alloc = [&](size_t bytes) {
    void* p = ws + off;
    off = (off + bytes + 255) & ~(size_t)255;
    return p;
  };
  unsigned short* Wqt = (unsigned short*)alloc((size_t)768 * 768 * 2);
  unsigned short* Wkt = (unsigned short*)alloc((size_t)768 * 1024 * 2);
  unsigned short* Wvt = (unsigned short*)alloc((size_t)768 * 1024 * 2);
  unsigned short* Wot = (unsigned short*)alloc((size_t)768 * 768 * 2);
  unsigned short* qbf = (unsigned short*)alloc((size_t)16 * 12 * 64 * 64 * 2);
  unsigned short* xbf = (unsigned short*)alloc((size_t)16 * 64 * 768 * 2);
  unsigned short* kbf = (unsigned short*)alloc((size_t)16 * 12 * 4096 * 64 * 2);
  unsigned short* vbf = (unsigned short*)alloc((size_t)16 * 12 * 4096 * 64 * 2);

  wconv_kernel<<<(768 * 768 + 255) / 256, 256, 0, stream>>>(Wq, Wqt, 768, 768);
  wconv_kernel<<<(1024 * 768 + 255) / 256, 256, 0, stream>>>(Wk, Wkt, 1024, 768);
  wconv_kernel<<<(1024 * 768 + 255) / 256, 256, 0, stream>>>(Wv, Wvt, 1024, 768);
  wconv_kernel<<<(768 * 768 + 255) / 256, 256, 0, stream>>>(Wo, Wot, 768, 768);

  qproj_kernel<<<192, 128, 0, stream>>>(query, Wqt, bq, qbf);
  kvproj_kernel<<<2048, 384, 0, stream>>>(kv, Wkt, bk, kbf, 0);
  kvproj_kernel<<<2048, 384, 0, stream>>>(kv, Wvt, bv, vbf, 1);
  attn_kernel<<<192, 128, 0, stream>>>(qbf, kbf, vbf, xbf);
  oproj_kernel<<<192, 128, 0, stream>>>(xbf, Wot, bo, out);
}
